// SoftModulizationHead_76467597738089
// MI455X (gfx1250) — matrix operand order repaired and checked by reading, not compiled, not run on hardware
//
#include <hip/hip_runtime.h>

typedef __attribute__((ext_vector_type(16))) _Float16 v16h;
typedef __attribute__((ext_vector_type(8)))  float    v8f;
typedef __attribute__((ext_vector_type(2)))  _Float16 h2;
typedef __attribute__((ext_vector_type(4)))  float    f4;
typedef __attribute__((ext_vector_type(4)))  int      i4;

#define DD 128
#define ASTR 272              // activation row stride (bytes): 16B aligned + bank-skewed
#define MIDSZ (16 * ASTR)     // 4352B: one [16 x 128] f16 activation buffer
#define PWSZ  23040           // per-wave LDS scratch
#define WA_BYTES 65536        // staged-weight region (2 x 32KB halves)
#define SMEM_BYTES (WA_BYTES + 8 * PWSZ)   // 249856 B < 320KB

// d_ws layout (bytes): prepped A-layout f16 weight blocks
#define WS_GATING 0          // 8t x 4c
#define WS_GW0    32768      // 1t x 4c
#define WS_COND1  36864      // 8t x 1c (K=16 padded to 32)
#define WS_GW1    45056
#define WS_COND2  49152      // 8t x 1c (K=32)
#define WS_GW2    57344
#define WS_PH1_I4 3840       // 61440/16
#define WS_CONDL  61440      // 8t x 2c (K=48 padded to 64)
#define WS_LAST   77824      // 1t x 4c (O=4 padded to 16)
#define WS_BASE   81920      // 16 x (8t x 4c) = 16 x 32768

// ---------------- CDNA5 async global->LDS copy (ASYNCcnt-tracked) -------------
static __device__ __forceinline__ void async_cp16(unsigned lds_off, const void* g) {
  asm volatile("global_load_async_to_lds_b128 %0, %1, off"
               :: "v"(lds_off), "v"((unsigned long long)(uintptr_t)g)
               : "memory");
}
static __device__ __forceinline__ void wait_async() {
  asm volatile("s_wait_asynccnt 0" ::: "memory");
}
static __device__ __forceinline__ unsigned lds_off32(const void* p) {
  return (unsigned)(uintptr_t)p;     // flat LDS address: low 32 bits = LDS offset
}

// ---------------- WMMA helper -------------------------------------------------
static __device__ __forceinline__ v8f wmma_f16(v16h a, v16h b, v8f c) {
  return __builtin_amdgcn_wmma_f32_16x16x32_f16(false, a, false, b, (short)0, c,
                                                false, false);
}

static __device__ __forceinline__ v16h pack16(const float* t) {
  v16h v;
#pragma unroll
  for (int e = 0; e < 16; ++e) v[e] = (_Float16)t[e];
  return v;
}

// A operand: contiguous 32B per lane per (tile,chunk) block
static __device__ __forceinline__ v16h loadA(const char* sm, int off, int blk, int lane) {
  return *(const v16h*)(sm + off + blk * 1024 + lane * 32);
}

// B operand from [batch][feat] f16 activation buffer: element e <-> k = 32c+16h+e
static __device__ __forceinline__ v16h loadBa(const char* sm, int aoff, int c, int n, int h) {
  const i4* p = (const i4*)(sm + aoff + n * ASTR + (c * 32 + h * 16) * 2);
  union { i4 u[2]; v16h v; } r; r.u[0] = p[0]; r.u[1] = p[1];
  return r.v;
}

// store C-layout f32 (8 tiles) as f16 into [batch][feat] buffer
static __device__ __forceinline__ void storeAct(char* sm, int aoff, const v8f* C,
                                                int n, int h, bool relu) {
#pragma unroll
  for (int t = 0; t < 8; ++t) {
    union { _Float16 hh[8]; i4 u; } P;
#pragma unroll
    for (int r = 0; r < 8; ++r) {
      float v = C[t][r]; if (relu) v = fmaxf(v, 0.f);
      P.hh[r] = (_Float16)v;
    }
    *(i4*)(sm + aoff + n * ASTR + (16 * t + 8 * h) * 2) = P.u;
  }
}

// B operand built from two flat-vector halves stored [o][batch] f32 in LDS
static __device__ __forceinline__ v16h buildBflat(const char* sm, int off0, bool r0,
                                                  int off1, bool r1, bool z1,
                                                  int n, int h) {
  float t[16];
  int base = h ? off1 : off0;
  bool rl = h ? r1 : r0;
#pragma unroll
  for (int e = 0; e < 16; ++e) {
    float v = *(const float*)(sm + base + (e * 16 + n) * 4);
    if (rl) v = fmaxf(v, 0.f);
    if (h && z1) v = 0.f;
    t[e] = v;
  }
  return pack16(t);
}

// B operand = relu( sum_m w[m] * mid_m ), fused mix on packed f16
static __device__ __forceinline__ v16h buildBmix(const char* sm, int midOff,
                                                 const float* wr, int c, int n, int h) {
  h2 acc[8];
#pragma unroll
  for (int k = 0; k < 8; ++k) { acc[k][0] = (_Float16)0; acc[k][1] = (_Float16)0; }
#pragma unroll
  for (int m = 0; m < 4; ++m) {
    const i4* p = (const i4*)(sm + midOff + m * MIDSZ + n * ASTR + (c * 32 + h * 16) * 2);
    union { i4 u[2]; h2 hh[8]; } mv; mv.u[0] = p[0]; mv.u[1] = p[1];
    _Float16 wf = (_Float16)wr[m];
    h2 wsp; wsp[0] = wf; wsp[1] = wf;
#pragma unroll
    for (int k = 0; k < 8; ++k) acc[k] += mv.hh[k] * wsp;
  }
  union { h2 hh[8]; v16h v; } R;
  h2 z; z[0] = (_Float16)0; z[1] = (_Float16)0;
#pragma unroll
  for (int k = 0; k < 8; ++k) R.hh[k] = __builtin_elementwise_max(acc[k], z);
  return R.v;
}

// C initializer from bias vector (feat = 16t+8h+r)
static __device__ __forceinline__ v8f bias128(const float* b, int f0) {
  f4 a = *(const f4*)(b + f0);
  f4 c = *(const f4*)(b + f0 + 4);
  v8f r;
#pragma unroll
  for (int u = 0; u < 4; ++u) { r[u] = a[u]; r[4 + u] = c[u]; }
  return r;
}

// softmax over groups of 4 along o=r+8h (within-lane), store w to [o][batch] f32
static __device__ __forceinline__ void softmaxStore(char* sm, int wOff, v8f R,
                                                    int n, int h) {
  float e[8];
#pragma unroll
  for (int r = 0; r < 8; ++r) e[r] = __expf(R[r]);
  float i0 = 1.f / (e[0] + e[1] + e[2] + e[3]);
  float i1 = 1.f / (e[4] + e[5] + e[6] + e[7]);
#pragma unroll
  for (int r = 0; r < 8; ++r) {
    float w = e[r] * (r < 4 ? i0 : i1);
    *(float*)(sm + wOff + ((r + 8 * h) * 16 + n) * 4) = w;
  }
}

static __device__ __forceinline__ void storeRaw(char* sm, int off, v8f R, int n, int h) {
#pragma unroll
  for (int r = 0; r < 8; ++r)
    *(float*)(sm + off + ((r + 8 * h) * 16 + n) * 4) = R[r];
}

// async-DMA module (idx+1)'s prepped weights into the other 32KB LDS half
static __device__ __forceinline__ void stageNext(char* sm, const char* ws, int i, int tid) {
  if (i < 15) {
    const char* s = ws + WS_BASE + (i + 1) * 32768 + tid * 16;
    const char* d = sm + (i & 1) * 32768 + tid * 16;
#pragma unroll
    for (int k = 0; k < 8; ++k)
      async_cp16(lds_off32(d + k * 4096), s + k * 4096);
  }
}

// ---------------- prep kernel: repack weights into WMMA A-layout f16 ----------
// logical A[m=out][k=in] from src stored [K_in][O_out]; zero-pad OOB.
__global__ void prep_kernel(const float* __restrict__ gat, const float* __restrict__ g0,
                            const float* __restrict__ c1,  const float* __restrict__ g1,
                            const float* __restrict__ c2,  const float* __restrict__ g2,
                            const float* __restrict__ cl,  const float* __restrict__ lst,
                            const float* __restrict__ bse, _Float16* __restrict__ dst) {
  int b = blockIdx.x, lane = threadIdx.x;
  const float* src; int K, O, Cc, matOff, bb;
  if      (b < 32) { src = gat; K = 128; O = 128; Cc = 4; matOff = WS_GATING; bb = b; }
  else if (b < 36) { src = g0;  K = 128; O = 16;  Cc = 4; matOff = WS_GW0;   bb = b - 32; }
  else if (b < 44) { src = c1;  K = 16;  O = 128; Cc = 1; matOff = WS_COND1; bb = b - 36; }
  else if (b < 48) { src = g1;  K = 128; O = 16;  Cc = 4; matOff = WS_GW1;   bb = b - 44; }
  else if (b < 56) { src = c2;  K = 32;  O = 128; Cc = 1; matOff = WS_COND2; bb = b - 48; }
  else if (b < 60) { src = g2;  K = 128; O = 16;  Cc = 4; matOff = WS_GW2;   bb = b - 56; }
  else if (b < 76) { src = cl;  K = 48;  O = 128; Cc = 2; matOff = WS_CONDL; bb = b - 60; }
  else if (b < 80) { src = lst; K = 128; O = 4;   Cc = 4; matOff = WS_LAST;  bb = b - 76; }
  else {
    int q = b - 80; int mi = q >> 5;
    src = bse + (size_t)mi * 128 * 128; K = 128; O = 128; Cc = 4;
    matOff = WS_BASE + mi * 32768; bb = q & 31;
  }
  int t = bb / Cc, c = bb % Cc;
  int m = t * 16 + (lane & 15);
  int h = lane >> 4;
  _Float16* d = dst + (matOff >> 1) + ((bb * 32 + lane) * 32) / 2;
#pragma unroll
  for (int e = 0; e < 16; ++e) {
    int k = c * 32 + h * 16 + e;   // contiguous k order (matches B operands)
    float v = (k < K && m < O) ? src[(size_t)k * O + m] : 0.f;
    d[e] = (_Float16)v;
  }
}

// ---------------- main fused kernel ------------------------------------------
__global__ void __launch_bounds__(256, 1)
soft_mod_main(const float* __restrict__ x, const int* __restrict__ task_id,
              const float* __restrict__ te_w, const float* __restrict__ te_b,
              const float* __restrict__ gat_b, const float* __restrict__ g0_b,
              const float* __restrict__ c1_b, const float* __restrict__ g1_b,
              const float* __restrict__ c2_b, const float* __restrict__ g2_b,
              const float* __restrict__ cl_b, const float* __restrict__ lst_b,
              const float* __restrict__ bse_b, const char* __restrict__ ws,
              float* __restrict__ out) {
  extern __shared__ char sm[];
  const int tid = threadIdx.x;
  const int wave = tid >> 5, lane = tid & 31;
  const int n = lane & 15, h = lane >> 4;
  const int row0 = (blockIdx.x * 8 + wave) * 16;
  const int pw = WA_BYTES + wave * PWSZ;
  const int midOff = pw;          // 4 x MIDSZ; mid[0] doubles as routing act buf
  const int actOff = pw;
  const int wbOff0 = pw + 17408, wbOff1 = pw + 18432, wbOff2 = pw + 19456;
  const int rawOff1 = pw + 20480, rawOff2 = pw + 21504;
  const int lwOff = pw + 22528;

  // ---- stage phase-1 routing weights (gating..gw2, 60KB) via async DMA ----
  for (int i = tid; i < WS_PH1_I4; i += 256)
    async_cp16(lds_off32(sm + i * 16), ws + i * 16);
  wait_async();
  __syncthreads();

  const int task = task_id[0];

  // ---- emb = relu((te_w[task]+te_b) * x); gating GEMM ----
  v16h xB[4];
  v8f G[8];
#pragma unroll
  for (int t = 0; t < 8; ++t) G[t] = bias128(gat_b, 16 * t + 8 * h);
#pragma unroll
  for (int c = 0; c < 4; ++c) {
    float xf[16], ef[16];
    const f4* xp = (const f4*)(x + (size_t)(row0 + n) * DD + c * 32 + h * 16);
    const f4* wp = (const f4*)(te_w + task * DD + c * 32 + h * 16);
    const f4* bp = (const f4*)(te_b + c * 32 + h * 16);
#pragma unroll
    for (int q = 0; q < 4; ++q) {
      f4 xv = xp[q], wv = wp[q], bv = bp[q];
#pragma unroll
      for (int u = 0; u < 4; ++u) {
        float xx = xv[u];
        xf[q * 4 + u] = xx;
        ef[q * 4 + u] = fmaxf((wv[u] + bv[u]) * xx, 0.f);
      }
    }
    xB[c] = pack16(xf);                 // raw x as B operand for base layer 0
    v16h Be = pack16(ef);
#pragma unroll
    for (int t = 0; t < 8; ++t)
      G[t] = wmma_f16(loadA(sm, WS_GATING, t * 4 + c, lane), Be, G[t]);
  }
  storeAct(sm, actOff, G, n, h, true);          // relu(G) -> act buffer

  // ---- gw_fc0 -> softmax -> w0 (== flats[0]) ----
  v8f R0 = bias128(g0_b, 8 * h);
#pragma unroll
  for (int c = 0; c < 4; ++c)
    R0 = wmma_f16(loadA(sm, WS_GW0, c, lane), loadBa(sm, actOff, c, n, h), R0);
  softmaxStore(sm, wbOff0, R0, n, h);

  // ---- cond_fc1 (K=16 padded) ----
  v8f CC[8];
#pragma unroll
  for (int t = 0; t < 8; ++t) CC[t] = bias128(c1_b, 16 * t + 8 * h);
  { v16h B1 = buildBflat(sm, wbOff0, false, wbOff0, false, true, n, h);
#pragma unroll
    for (int t = 0; t < 8; ++t)
      CC[t] = wmma_f16(loadA(sm, WS_COND1, t, lane), B1, CC[t]); }
#pragma unroll
  for (int t = 0; t < 8; ++t)
#pragma unroll
    for (int r = 0; r < 8; ++r) CC[t][r] = fmaxf(CC[t][r] * G[t][r], 0.f);
  storeAct(sm, actOff, CC, n, h, false);

  // ---- gw_fc1 -> raw1, w1 ----
  v8f R1 = bias128(g1_b, 8 * h);
#pragma unroll
  for (int c = 0; c < 4; ++c)
    R1 = wmma_f16(loadA(sm, WS_GW1, c, lane), loadBa(sm, actOff, c, n, h), R1);
  storeRaw(sm, rawOff1, R1, n, h);
  softmaxStore(sm, wbOff1, R1, n, h);

  // ---- cond_fc2 (K=32: [sm0 | relu(raw1)]) ----
#pragma unroll
  for (int t = 0; t < 8; ++t) CC[t] = bias128(c2_b, 16 * t + 8 * h);
  { v16h B2 = buildBflat(sm, wbOff0, false, rawOff1, true, false, n, h);
#pragma unroll
    for (int t = 0; t < 8; ++t)
      CC[t] = wmma_f16(loadA(sm, WS_COND2, t, lane), B2, CC[t]); }
#pragma unroll
  for (int t = 0; t < 8; ++t)
#pragma unroll
    for (int r = 0; r < 8; ++r) CC[t][r] = fmaxf(CC[t][r] * G[t][r], 0.f);
  storeAct(sm, actOff, CC, n, h, false);

  // ---- gw_fc2 -> raw2, w2 ----
  v8f R2 = bias128(g2_b, 8 * h);
#pragma unroll
  for (int c = 0; c < 4; ++c)
    R2 = wmma_f16(loadA(sm, WS_GW2, c, lane), loadBa(sm, actOff, c, n, h), R2);
  storeRaw(sm, rawOff2, R2, n, h);
  softmaxStore(sm, wbOff2, R2, n, h);

  __syncthreads();
  // ---- stage phase 2 via async DMA: cond_last+last @0, base module 0 @32KB ----
  for (int i = tid; i < 1280; i += 256)
    async_cp16(lds_off32(sm + i * 16), ws + WS_CONDL + i * 16);
  for (int i = tid; i < 2048; i += 256)
    async_cp16(lds_off32(sm + 32768 + i * 16), ws + WS_BASE + i * 16);
  wait_async();
  __syncthreads();

  // ---- cond_last (K=48 padded to 64) ----
#pragma unroll
  for (int t = 0; t < 8; ++t) CC[t] = bias128(cl_b, 16 * t + 8 * h);
  { v16h Bc0 = buildBflat(sm, wbOff0, false, rawOff1, true, false, n, h);
#pragma unroll
    for (int t = 0; t < 8; ++t)
      CC[t] = wmma_f16(loadA(sm, 0, t * 2 + 0, lane), Bc0, CC[t]); }
  { v16h Bc1 = buildBflat(sm, rawOff2, true, rawOff2, true, true, n, h);
#pragma unroll
    for (int t = 0; t < 8; ++t)
      CC[t] = wmma_f16(loadA(sm, 0, t * 2 + 1, lane), Bc1, CC[t]); }
#pragma unroll
  for (int t = 0; t < 8; ++t)
#pragma unroll
    for (int r = 0; r < 8; ++r) CC[t][r] = fmaxf(CC[t][r] * G[t][r], 0.f);
  storeAct(sm, actOff, CC, n, h, false);

  // ---- last_fc (O=4 padded) -> softmax -> last_w ----
  v8f LL;
  { f4 lb = *(const f4*)(lst_b);
#pragma unroll
    for (int r = 0; r < 8; ++r) {
      LL[r] = 0.f;
      if (h == 0 && r < 4) LL[r] = lb[r];
    } }
#pragma unroll
  for (int c = 0; c < 4; ++c)
    LL = wmma_f16(loadA(sm, 16384, c, lane), loadBa(sm, actOff, c, n, h), LL);
  { float e0 = __expf(LL[0]), e1 = __expf(LL[1]), e2 = __expf(LL[2]), e3 = __expf(LL[3]);
    float is = 1.f / (e0 + e1 + e2 + e3);
    if (h == 0) {
      *(float*)(sm + lwOff + (0 * 16 + n) * 4) = e0 * is;
      *(float*)(sm + lwOff + (1 * 16 + n) * 4) = e1 * is;
      *(float*)(sm + lwOff + (2 * 16 + n) * 4) = e2 * is;
      *(float*)(sm + lwOff + (3 * 16 + n) * 4) = e3 * is;
    } }
  float lwv[4];
#pragma unroll
  for (int j = 0; j < 4; ++j) lwv[j] = *(const float*)(sm + lwOff + (j * 16 + n) * 4);

  __syncthreads();   // all waves done reading phase-2 region before base staging

  // =====================  base network (fully unrolled)  =====================
  v8f OA[8];
#pragma unroll
  for (int t = 0; t < 8; ++t)
#pragma unroll
    for (int r = 0; r < 8; ++r) OA[t][r] = 0.f;

  // ---- layer 0: B = raw x (in registers), store mids in place ----
#pragma unroll
  for (int j = 0; j < 4; ++j) {
    stageNext(sm, ws, j, tid);
    v8f C[8];
#pragma unroll
    for (int t = 0; t < 8; ++t) C[t] = bias128(bse_b + j * DD, 16 * t + 8 * h);
    const int aoff = ((j & 1) ^ 1) * 32768;
#pragma unroll
    for (int c = 0; c < 4; ++c)
#pragma unroll
      for (int t = 0; t < 8; ++t)
        C[t] = wmma_f16(loadA(sm, aoff, t * 4 + c, lane), xB[c], C[t]);
    storeAct(sm, midOff + j * MIDSZ, C, n, h, false);
    wait_async();
    __syncthreads();
  }

  // ---- layers 1..2: prebuild all mixed B operands, then overwrite mids ----
  v16h Bop[4][4];
#pragma unroll
  for (int l = 1; l < 3; ++l) {
    const int wo = (l == 1) ? wbOff0 : wbOff1;
#pragma unroll
    for (int j = 0; j < 4; ++j) {
      float wr[4];
#pragma unroll
      for (int m = 0; m < 4; ++m)
        wr[m] = *(const float*)(sm + wo + ((j * 4 + m) * 16 + n) * 4);
#pragma unroll
      for (int c = 0; c < 4; ++c)
        Bop[j][c] = buildBmix(sm, midOff, wr, c, n, h);
    }
#pragma unroll
    for (int j = 0; j < 4; ++j) {
      const int i = l * 4 + j;
      stageNext(sm, ws, i, tid);
      v8f C[8];
#pragma unroll
      for (int t = 0; t < 8; ++t) C[t] = bias128(bse_b + i * DD, 16 * t + 8 * h);
      const int aoff = ((i & 1) ^ 1) * 32768;
#pragma unroll
      for (int c = 0; c < 4; ++c)
#pragma unroll
        for (int t = 0; t < 8; ++t)
          C[t] = wmma_f16(loadA(sm, aoff, t * 4 + c, lane), Bop[j][c], C[t]);
      storeAct(sm, midOff + j * MIDSZ, C, n, h, false);   // old mid j fully consumed
      wait_async();
      __syncthreads();
    }
  }

  // ---- layer 3: prebuild B, accumulate out = sum_j C_j * last_w[j] ----
#pragma unroll
  for (int j = 0; j < 4; ++j) {
    float wr[4];
#pragma unroll
    for (int m = 0; m < 4; ++m)
      wr[m] = *(const float*)(sm + wbOff2 + ((j * 4 + m) * 16 + n) * 4);
#pragma unroll
    for (int c = 0; c < 4; ++c)
      Bop[j][c] = buildBmix(sm, midOff, wr, c, n, h);
  }
#pragma unroll
  for (int j = 0; j < 4; ++j) {
    const int i = 12 + j;
    stageNext(sm, ws, i, tid);
    v8f C[8];
#pragma unroll
    for (int t = 0; t < 8; ++t) C[t] = bias128(bse_b + i * DD, 16 * t + 8 * h);
    const int aoff = ((i & 1) ^ 1) * 32768;
#pragma unroll
    for (int c = 0; c < 4; ++c)
#pragma unroll
      for (int t = 0; t < 8; ++t)
        C[t] = wmma_f16(loadA(sm, aoff, t * 4 + c, lane), Bop[j][c], C[t]);
    const float lw = lwv[j];
#pragma unroll
    for (int t = 0; t < 8; ++t)
#pragma unroll
      for (int r = 0; r < 8; ++r) OA[t][r] += C[t][r] * lw;
    wait_async();
    __syncthreads();
  }

  // ---- store output [B,128] f32 ----
  float* op = out + (size_t)(row0 + n) * DD;
#pragma unroll
  for (int t = 0; t < 8; ++t) {
    f4 lo, hi;
#pragma unroll
    for (int u = 0; u < 4; ++u) { lo[u] = OA[t][u]; hi[u] = OA[t][4 + u]; }
    *(f4*)(op + 16 * t + 8 * h) = lo;
    *(f4*)(op + 16 * t + 8 * h + 4) = hi;
  }
}

// ---------------- host launcher ----------------------------------------------
extern "C" void kernel_launch(void* const* d_in, const int* in_sizes, int n_in,
                              void* d_out, int out_size, void* d_ws, size_t ws_size,
                              hipStream_t stream) {
  const float* x   = (const float*)d_in[0];
  const int*   tid = (const int*)d_in[1];
  const float* tew = (const float*)d_in[2];
  const float* teb = (const float*)d_in[3];
  const float* gW  = (const float*)d_in[4];
  const float* gb  = (const float*)d_in[5];
  const float* g0W = (const float*)d_in[6];
  const float* g0b = (const float*)d_in[7];
  const float* c1W = (const float*)d_in[8];
  const float* c1b = (const float*)d_in[9];
  const float* g1W = (const float*)d_in[10];
  const float* g1b = (const float*)d_in[11];
  const float* c2W = (const float*)d_in[12];
  const float* c2b = (const float*)d_in[13];
  const float* g2W = (const float*)d_in[14];
  const float* g2b = (const float*)d_in[15];
  const float* clW = (const float*)d_in[16];
  const float* clb = (const float*)d_in[17];
  const float* lW  = (const float*)d_in[18];
  const float* lb  = (const float*)d_in[19];
  const float* bW  = (const float*)d_in[20];
  const float* bb  = (const float*)d_in[21];
  char* ws = (char*)d_ws;
  float* out = (float*)d_out;

  prep_kernel<<<592, 32, 0, stream>>>(gW, g0W, c1W, g1W, c2W, g2W, clW, lW, bW,
                                      (_Float16*)ws);

  (void)hipFuncSetAttribute((const void*)soft_mod_main,
                            hipFuncAttributeMaxDynamicSharedMemorySize, SMEM_BYTES);

  int Brows = in_sizes[0] / DD;          // 262144
  int blocks = Brows / 128;              // 8 waves x 16 rows per block
  soft_mod_main<<<blocks, 256, SMEM_BYTES, stream>>>(
      x, tid, tew, teb, gb, g0b, c1b, g1b, c2b, g2b, clb, lb, bb, ws, out);
}
